// ESN_50800873177277
// MI455X (gfx1250) — matrix operand order repaired and checked by reading, not compiled, not run on hardware
//
#include <hip/hip_runtime.h>
#include <math.h>

// ---- problem dims ----
#define B_  32
#define S_  1024
#define I_  128
#define H_  1024
#define O_  64
#define KF  1152   // fused K = H + I   ([W_res | W_in])
#define NWG 16     // persistent workgroups; each owns 64 H-columns
#define SLAB 64

typedef __attribute__((ext_vector_type(16))) _Float16 v16h;
typedef __attribute__((ext_vector_type(8)))  _Float16 v8h;
typedef __attribute__((ext_vector_type(8)))  float    v8f;

// Build a v16h WMMA fragment from two 16B-aligned 8-half chunks.
__device__ __forceinline__ v16h ld16(const _Float16* p0, const _Float16* p1) {
    v8h lo = *(const v8h*)p0;
    v8h hi = *(const v8h*)p1;
    return __builtin_shufflevector(lo, hi, 0,1,2,3,4,5,6,7,8,9,10,11,12,13,14,15);
}

// Branchless tanh: (e^2x - 1)/(e^2x + 1); exp2f -> single v_exp_f32 on amdgcn.
__device__ __forceinline__ float fast_tanh(float x) {
    float cx = fminf(fmaxf(x, -10.0f), 10.0f);
    float e  = exp2f(cx * 2.8853900817779268f);   // 2*log2(e)
    return (e - 1.0f) * __builtin_amdgcn_rcpf(e + 1.0f);
}

// ---- prep: fuse + convert weights to f16 [H][KF] ----
__global__ void __launch_bounds__(256) prep_w(const float* __restrict__ Wres,
                                              const float* __restrict__ Win,
                                              _Float16* __restrict__ Wf) {
    int idx = blockIdx.x * 256 + threadIdx.x;
    if (idx >= H_ * KF) return;
    int n = idx / KF, k = idx % KF;
    float v = (k < H_) ? Wres[n * H_ + k] : Win[n * I_ + (k - H_)];
    Wf[idx] = (_Float16)v;
}

// ---- prep: generic f32 -> f16 convert (x and W_out) ----
__global__ void __launch_bounds__(256) prep_f16(const float* __restrict__ src,
                                                _Float16* __restrict__ dst, int n) {
    int idx = blockIdx.x * 256 + threadIdx.x;
    if (idx < n) dst[idx] = (_Float16)src[idx];
}

// ---- persistent recurrence: 16 WGs x 256 threads (8 waves) ----
// Wave (nt, kh): nt = N tile (16 of 64 cols), kh = K half (18 of 36 k-steps).
// Weights live in 144 VGPRs per wave for the entire 1024-step scan: the only
// per-step memory traffic is h itself (64 KB through L0/L2) + an async-LDS
// prefetch of x_{t+1}. K halves are reduced through an 8 KB LDS buffer.
__global__ void __launch_bounds__(256, 1) esn_recur(
        const _Float16* __restrict__ Wf,   // [H][KF]
        const _Float16* __restrict__ x16,  // [B][S][I]
        _Float16* __restrict__ hs,         // [S+1][B][H], hs[0] pre-zeroed
        unsigned int* __restrict__ cnt) {  // monotonic step-barrier counter
    __shared__ _Float16 xbuf[2][B_ * I_];  // 16 KB: double-buffered x_t tile
    __shared__ float    redbuf[4][32][16]; // 8 KB: K-split reduction

    const int wg   = blockIdx.x;
    const int tid  = threadIdx.x;
    const int lane = tid & 31;
    const int w    = tid >> 5;
    const int l15  = lane & 15;
    const int hi   = lane >> 4;
    const int nt   = w & 3;   // N tile within the WG's 64-col slab
    const int kh   = w >> 2;  // K half (0: k-steps 0..17, 1: 18..35)

    // --- load this wave's 18 B fragments (weights) into registers ---
    v16h breg[18];
    {
        const _Float16* wrow =
            Wf + (size_t)(wg * SLAB + nt * 16 + l15) * KF + hi * 8;
        #pragma unroll
        for (int kk = 0; kk < 18; ++kk) {
            const int kg = kh * 18 + kk;
            breg[kk] = ld16(wrow + kg * 32, wrow + kg * 32 + 16);
        }
    }

    // --- async prefetch of x_0 into LDS buffer 0 (wave 0, ASYNCcnt) ---
    if (w == 0) {
        unsigned ldsb = (unsigned)(size_t)(&xbuf[0][0]);
        const _Float16* gx = x16 + (size_t)lane * S_ * I_;   // row b = lane
        #pragma unroll
        for (int j = 0; j < 16; ++j) {
            unsigned la = ldsb + (unsigned)(lane * I_ + j * 8) * 2;
            const _Float16* ga = gx + j * 8;
            asm volatile("global_load_async_to_lds_b128 %0, %1, off"
                         :: "v"(la), "v"(ga) : "memory");
        }
        asm volatile("s_wait_asynccnt 0x0" ::: "memory");
    }
    __syncthreads();

    const int ma0 = l15;       // A rows for accumulator c0 (batch 0..15)
    const int ma1 = 16 + l15;  // A rows for accumulator c1 (batch 16..31)

    for (int t = 0; t < S_; ++t) {
        // kick off async prefetch of x_{t+1} while this step computes
        if (w == 0 && t + 1 < S_) {
            unsigned ldsb = (unsigned)(size_t)(&xbuf[(t + 1) & 1][0]);
            const _Float16* gx = x16 + ((size_t)lane * S_ + (t + 1)) * I_;
            #pragma unroll
            for (int j = 0; j < 16; ++j) {
                unsigned la = ldsb + (unsigned)(lane * I_ + j * 8) * 2;
                const _Float16* ga = gx + j * 8;
                asm volatile("global_load_async_to_lds_b128 %0, %1, off"
                             :: "v"(la), "v"(ga) : "memory");
            }
        }

        const _Float16* h0 = hs + (size_t)t * B_ * H_ + (size_t)ma0 * H_ + hi * 8;
        const _Float16* h1 = hs + (size_t)t * B_ * H_ + (size_t)ma1 * H_ + hi * 8;
        const _Float16* xl0 = &xbuf[t & 1][ma0 * I_ + hi * 8];
        const _Float16* xl1 = &xbuf[t & 1][ma1 * I_ + hi * 8];
        v8f c0 = {}, c1 = {};

        if (kh == 0) {
            #pragma unroll
            for (int kk = 0; kk < 18; ++kk) {
                v16h a0 = ld16(h0 + kk * 32, h0 + kk * 32 + 16);
                v16h a1 = ld16(h1 + kk * 32, h1 + kk * 32 + 16);
                c0 = __builtin_amdgcn_wmma_f32_16x16x32_f16(false, a0, false, breg[kk],
                                                            (short)0, c0, false, false);
                c1 = __builtin_amdgcn_wmma_f32_16x16x32_f16(false, a1, false, breg[kk],
                                                            (short)0, c1, false, false);
            }
        } else {
            #pragma unroll
            for (int kk = 0; kk < 14; ++kk) {       // h k-steps 18..31
                const int kg = 18 + kk;
                v16h a0 = ld16(h0 + kg * 32, h0 + kg * 32 + 16);
                v16h a1 = ld16(h1 + kg * 32, h1 + kg * 32 + 16);
                c0 = __builtin_amdgcn_wmma_f32_16x16x32_f16(false, a0, false, breg[kk],
                                                            (short)0, c0, false, false);
                c1 = __builtin_amdgcn_wmma_f32_16x16x32_f16(false, a1, false, breg[kk],
                                                            (short)0, c1, false, false);
            }
            #pragma unroll
            for (int kk = 0; kk < 4; ++kk) {        // x k-steps 32..35 from LDS
                v16h a0 = ld16(xl0 + kk * 32, xl0 + kk * 32 + 16);
                v16h a1 = ld16(xl1 + kk * 32, xl1 + kk * 32 + 16);
                c0 = __builtin_amdgcn_wmma_f32_16x16x32_f16(false, a0, false, breg[14 + kk],
                                                            (short)0, c0, false, false);
                c1 = __builtin_amdgcn_wmma_f32_16x16x32_f16(false, a1, false, breg[14 + kk],
                                                            (short)0, c1, false, false);
            }
        }

        // --- reduce the two K halves through LDS ---
        if (kh == 1) {
            *(v8f*)&redbuf[nt][lane][0] = c0;
            *(v8f*)&redbuf[nt][lane][8] = c1;
        }
        __syncthreads();
        if (kh == 0) {
            v8f p0 = *(const v8f*)&redbuf[nt][lane][0];
            v8f p1 = *(const v8f*)&redbuf[nt][lane][8];
            c0 = c0 + p0;
            c1 = c1 + p1;
            // tanh + publish h_{t+1}; C layout: VGPR r -> M = 8*hi + r, lane -> N
            _Float16* hd = hs + (size_t)(t + 1) * B_ * H_
                              + (size_t)(wg * SLAB + nt * 16 + l15);
            #pragma unroll
            for (int r = 0; r < 8; ++r) {
                hd[(size_t)(hi * 8 + r) * H_]      = (_Float16)fast_tanh(c0[r]);
                hd[(size_t)(16 + hi * 8 + r) * H_] = (_Float16)fast_tanh(c1[r]);
            }
        }

        // --- device-wide step barrier ---
        __threadfence();
        __syncthreads();
        __builtin_amdgcn_s_cluster_barrier();  // NOP when unclustered
        if (tid == 0) {
            __hip_atomic_fetch_add(cnt, 1u, __ATOMIC_RELEASE, __HIP_MEMORY_SCOPE_AGENT);
            const unsigned target = (unsigned)(NWG * (t + 1));
            while (__hip_atomic_load(cnt, __ATOMIC_ACQUIRE, __HIP_MEMORY_SCOPE_AGENT) < target) {
                __builtin_amdgcn_s_sleep(2);
            }
        }
        if (w == 0) {  // x_{t+1} must be resident before next step reads it
            asm volatile("s_wait_asynccnt 0x0" ::: "memory");
        }
        __syncthreads();
    }
}

// ---- readout: out[b,s,o] = hs[s+1,b,:] . Wout[o,:] + bias[o] ----
// 8192 16x16 tiles over [S*B=32768] x [O=64], K = H = 1024.
__global__ void __launch_bounds__(256) esn_readout(
        const _Float16* __restrict__ hs,    // [S+1][B][H]
        const _Float16* __restrict__ Wo16,  // [O][H]
        const float*    __restrict__ bias,  // [O]
        float* __restrict__ out) {          // [B][S][O]
    const int tid  = threadIdx.x;
    const int lane = tid & 31;
    const int w    = tid >> 5;
    const int l15  = lane & 15;
    const int hi   = lane >> 4;
    const int gw   = blockIdx.x * 8 + w;   // global tile id
    const int nt   = gw & 3;               // O tile
    const int mtg  = gw >> 2;              // row tile in (t*32+b) space
    const int row  = mtg * 16 + l15;
    const int t    = row >> 5, b = row & 31;
    const int o    = nt * 16 + l15;
    const _Float16* arow = hs   + ((size_t)(t + 1) * B_ + b) * H_;
    const _Float16* brow = Wo16 + (size_t)o * H_;
    v8f c = {};
    #pragma unroll 4
    for (int kk = 0; kk < 32; ++kk) {
        const int ka = kk * 32 + hi * 8;
        v16h a = ld16(arow + ka, arow + ka + 16);
        const int kb = kk * 32 + hi * 8;
        v16h bf = ld16(brow + kb, brow + kb + 16);
        c = __builtin_amdgcn_wmma_f32_16x16x32_f16(false, a, false, bf,
                                                   (short)0, c, false, false);
    }
    const float bo = bias[o];
    #pragma unroll
    for (int r = 0; r < 8; ++r) {
        const int rr = mtg * 16 + hi * 8 + r;   // row for this C element
        const int tt = rr >> 5, bb = rr & 31;
        out[((size_t)bb * S_ + tt) * O_ + o] = c[r] + bo;
    }
}

extern "C" void kernel_launch(void* const* d_in, const int* in_sizes, int n_in,
                              void* d_out, int out_size, void* d_ws, size_t ws_size,
                              hipStream_t stream) {
    (void)in_sizes; (void)n_in; (void)out_size; (void)ws_size;
    const float* x    = (const float*)d_in[0];   // [B,S,I]
    const float* Wres = (const float*)d_in[1];   // [H,H]
    const float* Win  = (const float*)d_in[2];   // [H,I]
    const float* Wow  = (const float*)d_in[3];   // [O,H]
    const float* Wob  = (const float*)d_in[4];   // [O]

    char* ws = (char*)d_ws;
    size_t off = 0;
    unsigned int* cnt  = (unsigned int*)(ws + off); off += 256;
    _Float16* Wf   = (_Float16*)(ws + off); off += (size_t)H_ * KF * 2;            // 2.3 MB
    _Float16* Wo16 = (_Float16*)(ws + off); off += (size_t)O_ * H_ * 2;            // 128 KB
    _Float16* x16  = (_Float16*)(ws + off); off += (size_t)B_ * S_ * I_ * 2;       // 8 MB
    _Float16* hs   = (_Float16*)(ws + off); off += (size_t)(S_ + 1) * B_ * H_ * 2; // 67 MB

    hipMemsetAsync(cnt, 0, 256, stream);                   // barrier counter
    hipMemsetAsync(hs, 0, (size_t)B_ * H_ * 2, stream);    // h0 = 0

    prep_w  <<<(H_ * KF + 255) / 256, 256, 0, stream>>>(Wres, Win, Wf);
    prep_f16<<<(O_ * H_ + 255) / 256, 256, 0, stream>>>(Wow, Wo16, O_ * H_);
    prep_f16<<<(B_ * S_ * I_ + 255) / 256, 256, 0, stream>>>(x, x16, B_ * S_ * I_);

    esn_recur<<<NWG, 256, 0, stream>>>(Wf, x16, hs, cnt);

    esn_readout<<<(2048 * 4) / 8, 256, 0, stream>>>(hs, Wo16, Wob, (float*)d_out);
}
